// BiLSTM_CRF_63024350101772
// MI455X (gfx1250) — matrix operand order repaired and checked by reading, not compiled, not run on hardware
//
#include <hip/hip_runtime.h>
#include <hip/hip_bf16.h>

typedef __attribute__((ext_vector_type(16))) _Float16 v16h;
typedef __attribute__((ext_vector_type(8)))  _Float16 v8h;
typedef __attribute__((ext_vector_type(8)))  float    v8f;
typedef __attribute__((ext_vector_type(4)))  float    v4f;

#define BB 256          // batch
#define SS 512          // seq len
#define EE 300          // embed dim
#define HH 75           // hidden per dir
#define G4 300          // 4*H
#define NPAD 304        // padded output cols (19 tiles of 16)
#define KIH 320         // padded K for input GEMM (10 chunks of 32)
#define KHH 96          // padded K for recurrent GEMM (3 chunks of 32)
#define TT 6
#define START_TAG 4
#define STOP_TAG 5

// ---------------------------------------------------------------------------
// Kernel 1: weight conversion to f16, gate-interleaved reorder (n = 4*j + g),
// bias fuse (b_ih + b_hh).  Zero padding everywhere.
// ---------------------------------------------------------------------------
__global__ void prep_kernel(const float* wihf, const float* wihb,
                            const float* whhf, const float* whhb,
                            const float* bihf, const float* bhhf,
                            const float* bihb, const float* bhhb,
                            _Float16* wih16, _Float16* whh16, float* biasR) {
  int idx = blockIdx.x * blockDim.x + threadIdx.x;
  const int NIH = 2 * NPAD * KIH;
  const int NHH = 2 * NPAD * KHH;
  const int NB  = 2 * NPAD;
  if (idx < NIH) {
    int dir = idx / (NPAD * KIH);
    int rem = idx - dir * (NPAD * KIH);
    int n = rem / KIH, k = rem - n * KIH;
    float v = 0.f;
    if (n < G4 && k < EE) {
      int j = n >> 2, g = n & 3;
      const float* w = dir ? wihb : wihf;
      v = w[(g * HH + j) * EE + k];
    }
    wih16[idx] = (_Float16)v;
    return;
  }
  idx -= NIH;
  if (idx < NHH) {
    int dir = idx / (NPAD * KHH);
    int rem = idx - dir * (NPAD * KHH);
    int n = rem / KHH, k = rem - n * KHH;
    float v = 0.f;
    if (n < G4 && k < HH) {
      int j = n >> 2, g = n & 3;
      const float* w = dir ? whhb : whhf;
      v = w[(g * HH + j) * HH + k];
    }
    whh16[idx] = (_Float16)v;
    return;
  }
  idx -= NHH;
  if (idx < NB) {
    int dir = idx / NPAD;
    int n = idx - dir * NPAD;
    float v = 0.f;
    if (n < G4) {
      int j = n >> 2, g = n & 3;
      v = dir ? (bihb[g * HH + j] + bhhb[g * HH + j])
              : (bihf[g * HH + j] + bhhf[g * HH + j]);
    }
    biasR[idx] = v;
  }
}

// ---------------------------------------------------------------------------
// Kernel 2: big input-gate GEMM.  One wave = one 16-row M tile x one
// direction; A (gathered embedding rows -> f16) held in VGPRs across all 19
// N tiles; 10 x v_wmma_f32_16x16x32_f16 per tile.
// xg layout: [dir][s][b][NPAD] f32 (gate-interleaved cols, bias included).
// ---------------------------------------------------------------------------
__global__ void xg_kernel(const int* __restrict__ sentence,
                          const float* __restrict__ embedding,
                          const _Float16* __restrict__ wih16,
                          const float* __restrict__ biasR,
                          float* __restrict__ xg) {
  int wid  = (blockIdx.x * blockDim.x + threadIdx.x) >> 5;
  int lane = threadIdx.x & 31;
  int mtile = wid & 8191;     // 131072/16 M tiles
  int dir   = wid >> 13;
  int sel = lane >> 4;        // lane half
  int lm  = lane & 15;
  int r = mtile * 16 + lm;    // global row for A (r = s*B + b)
  int s = r >> 8;
  int b = r & 255;
  int tok = sentence[b * SS + s];
  const float* emb = embedding + (size_t)tok * EE;

  // Preload A: 10 K-chunks of 32, contiguous k order (k = 16*sel + e)
  v16h A[10];
#pragma unroll
  for (int kk = 0; kk < 10; ++kk) {
    int k0 = kk * 32;
    v16h a;
#pragma unroll
    for (int e = 0; e < 16; ++e) {
      int k = k0 + sel * 16 + e;
      float v = (k < EE) ? emb[k] : 0.f;
      a[e] = (_Float16)v;
    }
    A[kk] = a;
  }

  const _Float16* wb = wih16 + (size_t)dir * (NPAD * KIH);
  float* xgo = xg + (size_t)dir * ((size_t)SS * BB * NPAD);

  for (int nt = 0; nt < 19; ++nt) {
    int n = nt * 16 + lm;     // B col for this lane
    v8f acc = {};
#pragma unroll
    for (int kk = 0; kk < 10; ++kk) {
      v16h bmat = *(const v16h*)(wb + (size_t)n * KIH + kk * 32 + sel * 16);
      acc = __builtin_amdgcn_wmma_f32_16x16x32_f16(
          false, A[kk], false, bmat, (short)0, acc, false, false);
    }
    float bias = biasR[dir * NPAD + n];
    size_t base = (size_t)(mtile * 16) * NPAD + n;
#pragma unroll
    for (int i = 0; i < 8; ++i) {
      int row = i + sel * 8;
      xgo[base + (size_t)row * NPAD] = acc[i] + bias;
    }
  }
}

// ---------------------------------------------------------------------------
// Kernel 3: LSTM recurrence.  Batch rows are independent, so: one block per
// (direction, 16-row batch tile) = 32 blocks of 128 threads (4 waves).
// h resident in LDS as f16 (WMMA A operand), c resident in LDS f32.
// Per step: gates = xg[s] + h @ W_hh^T via WMMA (4-way N split across waves),
// then a per-wave LDS transpose (stride-20 rows -> conflict-free b128 reads)
// hands every lane two complete (row,unit) i/f/g/o quadruples so the
// sigmoid/tanh block runs at full EXEC.
// ---------------------------------------------------------------------------
__global__ void __launch_bounds__(128)
lstm_kernel(const float* __restrict__ h0,
            const float* __restrict__ c0,
            const _Float16* __restrict__ whh16,
            const float* __restrict__ xg,
            float* __restrict__ hs) {
  int blk = blockIdx.x;            // 0..31
  int dir = blk >> 4;
  int mtile = blk & 15;            // 16-row batch tile
  int m0 = mtile * 16;

  __shared__ __align__(32) _Float16 hbuf[16 * KHH];   // 3 KB
  __shared__ float cbuf[16 * HH];                      // 4.8 KB
  __shared__ float sbuf[4][16 * 20];                   // 5 KB transpose scratch

  int tid = threadIdx.x;
  for (int i = tid; i < 16 * KHH; i += 128) {
    int r = i / KHH, k = i - r * KHH;
    hbuf[i] = (_Float16)((k < HH) ? h0[dir * BB * HH + (m0 + r) * HH + k] : 0.f);
  }
  for (int i = tid; i < 16 * HH; i += 128) {
    cbuf[i] = c0[dir * BB * HH + m0 * HH + i];
  }
  __syncthreads();

  int wv = tid >> 5;               // 0..3 : N-split
  int lane = tid & 31;
  int sel = lane >> 4, lm = lane & 15;
  // fixed (row, unit-slot) pair assignment for the activation phase
  int tr_r = lane & 15;            // local batch row
  int tr_u0 = lane >> 4;           // unit slot for rep 0 (rep 1 adds 2)

  const _Float16* wdir = whh16 + (size_t)dir * (NPAD * KHH);
  const float* xdir = xg + (size_t)dir * ((size_t)SS * BB * NPAD);
  float* hdir = hs + (size_t)dir * ((size_t)SS * BB * HH);

  for (int step = 0; step < SS; ++step) {
    int s = dir ? (SS - 1 - step) : step;
    // ---- A preload (old h) from LDS, contiguous k order (k = 16*sel + e) ----
    v16h A[3];
#pragma unroll
    for (int kk = 0; kk < 3; ++kk) {
      int k0 = kk * 32;
      v8h lo = *(const v8h*)(hbuf + lm * KHH + k0 + sel * 16);
      v8h hi = *(const v8h*)(hbuf + lm * KHH + k0 + sel * 16 + 8);
      v16h a;
#pragma unroll
      for (int e = 0; e < 8; ++e) { a[e] = lo[e]; a[e + 8] = hi[e]; }
      A[kk] = a;
    }
    __syncthreads();   // all waves have consumed old h

    const float* xrow = xdir + (size_t)s * (BB * NPAD) + (size_t)m0 * NPAD;

    // prefetch next step's xg slab for this block (19.5 KB)
    {
      int sn = dir ? (s - 1) : (s + 1);
      if (sn >= 0 && sn < SS) {
        const char* p =
            (const char*)(xdir + (size_t)sn * (BB * NPAD) + (size_t)m0 * NPAD);
        for (int o = tid * 128; o < 16 * NPAD * 4; o += 128 * 128)
          __builtin_prefetch(p + o, 0, 1);
      }
    }

    for (int nt = wv; nt < 19; nt += 4) {
      int n = nt * 16 + lm;
      v8f acc;
#pragma unroll
      for (int i = 0; i < 8; ++i)
        acc[i] = xrow[(size_t)(i + sel * 8) * NPAD + n];
#pragma unroll
      for (int kk = 0; kk < 3; ++kk) {
        v16h bmat = *(const v16h*)(wdir + (size_t)n * KHH + kk * 32 + sel * 16);
        acc = __builtin_amdgcn_wmma_f32_16x16x32_f16(
            false, A[kk], false, bmat, (short)0, acc, false, false);
      }
      // per-wave LDS transpose: tile rows with stride 20 (conflict-free)
#pragma unroll
      for (int i = 0; i < 8; ++i)
        sbuf[wv][(i + sel * 8) * 20 + lm] = acc[i];
      // each lane: 2 (row, unit) pairs, 4 gates contiguous -> ds_load_b128
#pragma unroll
      for (int rep = 0; rep < 2; ++rep) {
        int u = tr_u0 + 2 * rep;         // unit slot 0..3 within tile
        int j = nt * 4 + u;              // hidden unit
        v4f g = *(const v4f*)&sbuf[wv][tr_r * 20 + u * 4];
        if (j < HH) {
          float ig = 1.f / (1.f + __expf(-g[0]));
          float fg = 1.f / (1.f + __expf(-g[1]));
          float gg = tanhf(g[2]);
          float og = 1.f / (1.f + __expf(-g[3]));
          float c = fg * cbuf[tr_r * HH + j] + ig * gg;
          cbuf[tr_r * HH + j] = c;
          float h = og * tanhf(c);
          hbuf[tr_r * KHH + j] = (_Float16)h;
          hdir[(size_t)s * (BB * HH) + (m0 + tr_r) * HH + j] = h;
        }
      }
    }
    __syncthreads();   // new h visible to all waves
  }
}

// ---------------------------------------------------------------------------
// Kernel 4: feats = [hf|hb] @ w_out^T + b_out   (K=150, N=6 -> VALU)
// ---------------------------------------------------------------------------
__global__ void feats_kernel(const float* __restrict__ hs,
                             const float* __restrict__ w_out,
                             const float* __restrict__ b_out,
                             float* __restrict__ feats) {
  int idx = blockIdx.x * blockDim.x + threadIdx.x;   // s*256 + b
  if (idx >= SS * BB) return;
  const float* hf = hs + (size_t)idx * HH;
  const float* hb = hs + (size_t)SS * BB * HH + (size_t)idx * HH;
  float out[TT];
#pragma unroll
  for (int t = 0; t < TT; ++t) out[t] = b_out[t];
  for (int k = 0; k < HH; ++k) {
    float a = hf[k], c = hb[k];
#pragma unroll
    for (int t = 0; t < TT; ++t)
      out[t] += a * w_out[t * (2 * HH) + k] + c * w_out[t * (2 * HH) + HH + k];
  }
  float* fo = feats + (size_t)idx * TT;
#pragma unroll
  for (int t = 0; t < TT; ++t) fo[t] = out[t];
}

// ---------------------------------------------------------------------------
// Kernel 5: CRF.  6 lanes cooperate per batch row (8-lane groups, 4 groups
// per wave): lane j owns alpha[j] and one row of transitions in registers;
// alpha broadcast via __shfl; one 6-way logsumexp per lane per step.
// 64 waves over 8 blocks.
// ---------------------------------------------------------------------------
__global__ void __launch_bounds__(256)
crf_kernel(const float* __restrict__ feats,
           const int* __restrict__ tags,
           const float* __restrict__ transitions,
           float* __restrict__ out) {
  __shared__ float tr[TT * TT];
  int tid = threadIdx.x;
  if (tid < TT * TT) tr[tid] = transitions[tid];
  __syncthreads();

  int lane = tid & 31;
  int grp = lane >> 3;                       // 4 batch rows per wave
  int j = lane & 7;                          // tag index (active j < 6)
  int qb = lane & ~7;
  int gw = (blockIdx.x * blockDim.x + tid) >> 5;   // global wave 0..63
  int b = gw * 4 + grp;

  int jc = (j < TT) ? j : 0;
  float trj[TT];
#pragma unroll
  for (int i = 0; i < TT; ++i) trj[i] = tr[jc * TT + i];

  float alpha = (j == START_TAG) ? 0.f : -10000.f;
  float gold = 0.f;
  int prev = START_TAG;

  for (int s = 0; s < SS; ++s) {
    const float* f = feats + ((size_t)s * BB + b) * TT;
    float av[TT];
#pragma unroll
    for (int i = 0; i < TT; ++i) av[i] = __shfl(alpha, qb + i, 32);
    if (j < TT) {
      float fj = f[j];
      float m = av[0] + trj[0];
#pragma unroll
      for (int i = 1; i < TT; ++i) m = fmaxf(m, av[i] + trj[i]);
      float sum = 0.f;
#pragma unroll
      for (int i = 0; i < TT; ++i) sum += __expf(av[i] + trj[i] - m);
      alpha = fj + m + __logf(sum);
    }
    if (j == 0) {
      int tg = tags[b * SS + s];
      gold += tr[tg * TT + prev] + f[tg];
      prev = tg;
    }
  }

  float av[TT];
#pragma unroll
  for (int i = 0; i < TT; ++i) av[i] = __shfl(alpha, qb + i, 32);
  if (j == 0) {
    gold += tr[STOP_TAG * TT + prev];
    float m = av[0] + tr[STOP_TAG * TT + 0];
#pragma unroll
    for (int i = 1; i < TT; ++i) m = fmaxf(m, av[i] + tr[STOP_TAG * TT + i]);
    float sum = 0.f;
#pragma unroll
    for (int i = 0; i < TT; ++i) sum += __expf(av[i] + tr[STOP_TAG * TT + i] - m);
    out[b] = (m + __logf(sum)) - gold;
  }
}

// ---------------------------------------------------------------------------
extern "C" void kernel_launch(void* const* d_in, const int* in_sizes, int n_in,
                              void* d_out, int out_size, void* d_ws, size_t ws_size,
                              hipStream_t stream) {
  (void)in_sizes; (void)n_in; (void)out_size; (void)ws_size;
  const int*   sentence    = (const int*)d_in[0];
  const int*   tags        = (const int*)d_in[1];
  const float* embedding   = (const float*)d_in[2];
  const float* w_ih_f      = (const float*)d_in[3];
  const float* w_hh_f      = (const float*)d_in[4];
  const float* b_ih_f      = (const float*)d_in[5];
  const float* b_hh_f      = (const float*)d_in[6];
  const float* w_ih_b      = (const float*)d_in[7];
  const float* w_hh_b      = (const float*)d_in[8];
  const float* b_ih_b      = (const float*)d_in[9];
  const float* b_hh_b      = (const float*)d_in[10];
  const float* h0          = (const float*)d_in[11];
  const float* c0          = (const float*)d_in[12];
  const float* w_out       = (const float*)d_in[13];
  const float* b_out       = (const float*)d_in[14];
  const float* transitions = (const float*)d_in[15];
  float* out = (float*)d_out;

  char* ws = (char*)d_ws;
  size_t off = 0;
  auto take = [&](size_t bytes) {
    size_t r = off;
    off += (bytes + 255) & ~(size_t)255;
    return r;
  };
  _Float16* wih16 = (_Float16*)(ws + take((size_t)2 * NPAD * KIH * 2));
  _Float16* whh16 = (_Float16*)(ws + take((size_t)2 * NPAD * KHH * 2));
  float*    biasR = (float*)   (ws + take((size_t)2 * NPAD * 4));
  float*    xgbuf = (float*)   (ws + take((size_t)2 * SS * BB * NPAD * 4));
  float*    hsbuf = (float*)   (ws + take((size_t)2 * SS * BB * HH * 4));
  float*    fbuf  = (float*)   (ws + take((size_t)SS * BB * TT * 4));

  {
    int total = 2 * NPAD * KIH + 2 * NPAD * KHH + 2 * NPAD;
    prep_kernel<<<(total + 255) / 256, 256, 0, stream>>>(
        w_ih_f, w_ih_b, w_hh_f, w_hh_b, b_ih_f, b_hh_f, b_ih_b, b_hh_b,
        wih16, whh16, biasR);
  }
  // 16384 waves (8192 M tiles x 2 dirs), 8 waves per block
  xg_kernel<<<2048, 256, 0, stream>>>(sentence, embedding, wih16, biasR, xgbuf);
  lstm_kernel<<<32, 128, 0, stream>>>(h0, c0, whh16, xgbuf, hsbuf);
  feats_kernel<<<(SS * BB) / 256, 256, 0, stream>>>(hsbuf, w_out, b_out, fbuf);
  crf_kernel<<<8, 256, 0, stream>>>(fbuf, tags, transitions, out);
}
